// RelationNetwork_68418829025626
// MI455X (gfx1250) — matrix operand order repaired and checked by reading, not compiled, not run on hardware
//
#include <hip/hip_runtime.h>

// ---------------------------------------------------------------------------
// CDNA5 (gfx1250) Relation Network, wave32 + v_wmma_f32_16x16x32_bf16
// ---------------------------------------------------------------------------

typedef __attribute__((ext_vector_type(16))) __bf16 v16bf;
typedef __attribute__((ext_vector_type(8)))  float  v8f;

union ABFrag { uint4 u[2]; v16bf v; };

#define INPUT_DIM 1024
#define HIDDEN    512
#define NUM_WAYS  64
#define NUM_SUP   1024
#define NUM_QRY   4096

// fp32 -> bf16, round-to-nearest-even
__device__ __forceinline__ unsigned short f2bf(float x) {
    unsigned int u = __float_as_uint(x);
    u += 0x7FFFu + ((u >> 16) & 1u);
    return (unsigned short)(u >> 16);
}
__device__ __forceinline__ float b2f(unsigned short s) {
    return __uint_as_float(((unsigned int)s) << 16);
}

// A fragment (16x32 bf16, M x K): lane = {row m = lane&15, half = lane>>4}
//   elements 0..7  : K = k0 + 8*half + e        (contiguous 16B run)
//   elements 8..15 : K = k0 + 16 + 8*half + e-8 (contiguous 16B run)
__device__ __forceinline__ v16bf load_frag_a(const unsigned short* __restrict__ A,
                                             int lda, int m0, int k0, int lane) {
    const int m = lane & 15, hf = lane >> 4;
    const unsigned short* p = A + (size_t)(m0 + m) * lda + k0 + 8 * hf;
    ABFrag f;
    f.u[0] = *(const uint4*)(p);
    f.u[1] = *(const uint4*)(p + 16);
    return f.v;
}

// B fragment (32x16 bf16, K x N) from transposed weight BT[N][K]:
//   lane = {col n = lane&15, half = lane>>4}; elements 0..7: K = k0 + 8*half + e,
//   elements 8..15: K = k0 + 16 + 8*half + e-8  => two contiguous 16B runs per lane
__device__ __forceinline__ v16bf load_frag_b(const unsigned short* __restrict__ BT,
                                             int K, int n0, int k0, int lane) {
    const int n = lane & 15, hf = lane >> 4;
    const unsigned short* p = BT + (size_t)(n0 + n) * K + k0 + 8 * hf;
    ABFrag f;
    f.u[0] = *(const uint4*)(p);
    f.u[1] = *(const uint4*)(p + 16);
    return f.v;
}

// ---------------------------------------------------------------------------
// Prep kernels
// ---------------------------------------------------------------------------
__global__ __launch_bounds__(256) void cvt_f32_to_bf16(const float* __restrict__ in,
                                                       unsigned short* __restrict__ out,
                                                       int n) {
    int i = blockIdx.x * 256 + threadIdx.x;
    if (i < n) out[i] = f2bf(in[i]);
}

// out[n*K + k] = bf16(in[k*N + n])   (weight [K,N] row-major -> [N,K] bf16)
__global__ __launch_bounds__(256) void transpose_cvt(const float* __restrict__ in,
                                                     unsigned short* __restrict__ out,
                                                     int K, int N) {
    int idx = blockIdx.x * 256 + threadIdx.x;
    if (idx < K * N) {
        int n = idx / K;
        int k = idx - n * K;
        out[idx] = f2bf(in[(size_t)k * N + n]);
    }
}

// ---------------------------------------------------------------------------
// Generic WMMA GEMM: C[M,512] = act(A[M,K] @ B[K,512] + bias)
// A: bf16 row-major (lda=K). BT: bf16 [512,K]. One block = 32 rows, 8 waves
// each own a 64-wide N strip (2 M-subtiles x 4 N-subtiles => 2x B reuse).
// Grid = M/32 blocks.
// ---------------------------------------------------------------------------
template <bool RELU, bool OUTBF16>
__global__ __launch_bounds__(256) void gemm512(const unsigned short* __restrict__ A,
                                               const unsigned short* __restrict__ BT,
                                               const float* __restrict__ bias,
                                               void* __restrict__ Cout, int K) {
    const int tid  = threadIdx.x;
    const int wave = tid >> 5;
    const int lane = tid & 31;
    const int m0   = blockIdx.x * 32;
    const int n0w  = wave * 64;
    const int N    = 512;

    v8f acc[2][4];
#pragma unroll
    for (int mt = 0; mt < 2; ++mt)
#pragma unroll
        for (int t = 0; t < 4; ++t) acc[mt][t] = v8f{};

    for (int k0 = 0; k0 < K; k0 += 32) {
        v16bf a[2];
#pragma unroll
        for (int mt = 0; mt < 2; ++mt)
            a[mt] = load_frag_a(A, K, m0 + 16 * mt, k0, lane);
#pragma unroll
        for (int t = 0; t < 4; ++t) {
            v16bf b = load_frag_b(BT, K, n0w + 16 * t, k0, lane);
#pragma unroll
            for (int mt = 0; mt < 2; ++mt)
                acc[mt][t] = __builtin_amdgcn_wmma_f32_16x16x32_bf16(
                    false, a[mt], false, b, (short)0, acc[mt][t], false, false);
        }
    }

    const int nl = lane & 15, hf = lane >> 4;
#pragma unroll
    for (int t = 0; t < 4; ++t) {
        const int n = n0w + 16 * t + nl;
        const float bv = bias ? bias[n] : 0.0f;
#pragma unroll
        for (int mt = 0; mt < 2; ++mt) {
#pragma unroll
            for (int r = 0; r < 8; ++r) {
                float v = acc[mt][t][r] + bv;
                if (RELU) v = fmaxf(v, 0.0f);
                const size_t o = (size_t)(m0 + 16 * mt + r + 8 * hf) * N + n;
                if constexpr (OUTBF16) ((unsigned short*)Cout)[o] = f2bf(v);
                else                   ((float*)Cout)[o] = v;
            }
        }
    }
}

// ---------------------------------------------------------------------------
// Class prototypes: segment mean over support features (deterministic)
// One block per class; thread owns columns tid and tid+256.
// ---------------------------------------------------------------------------
__global__ __launch_bounds__(256) void proto_kernel(const unsigned short* __restrict__ sf,
                                                    const int* __restrict__ y,
                                                    float* __restrict__ protos,
                                                    unsigned short* __restrict__ protosb) {
    const int c = blockIdx.x, tid = threadIdx.x;
    float a0 = 0.f, a1 = 0.f;
    int cnt = 0;
    for (int s = 0; s < NUM_SUP; ++s) {
        if (y[s] == c) {
            ++cnt;
            a0 += b2f(sf[(size_t)s * HIDDEN + tid]);
            a1 += b2f(sf[(size_t)s * HIDDEN + tid + 256]);
        }
    }
    const float d = fmaxf((float)cnt, 1.0f);
    a0 /= d; a1 /= d;
    protos [(size_t)c * HIDDEN + tid]       = a0;
    protos [(size_t)c * HIDDEN + tid + 256] = a1;
    protosb[(size_t)c * HIDDEN + tid]       = f2bf(a0);
    protosb[(size_t)c * HIDDEN + tid + 256] = f2bf(a1);
}

// ---------------------------------------------------------------------------
// Fused relation kernel.
// Block = (64 query rows) x (1 way); 8 waves.
//  1) h = relu(q_part + p_part) staged in LDS as bf16 [64][520] (padded pitch:
//     1040 B = 260 dwords == 4 mod 64 banks -> conflict-spread ds_read_b128)
//  2) each wave: 64-wide N strip of h @ R2w via WMMA; 4 M-subtiles x 4
//     N-subtiles => 4x reuse of every B fragment (1 b128 load per WMMA)
//  3) fused bias+ReLU+R3w dot, shuffle + LDS reduce, sigmoid, write [Q,W]
// ---------------------------------------------------------------------------
__global__ __launch_bounds__(256) void relation_fused(const float* __restrict__ qpart,
                                                      const float* __restrict__ ppart,
                                                      const unsigned short* __restrict__ R2wT,
                                                      const float* __restrict__ R2b,
                                                      const float* __restrict__ R3w,
                                                      const float* __restrict__ R3b,
                                                      float* __restrict__ out) {
    constexpr int H = HIDDEN, LDH = H + 8;  // pitch 520 shorts = 1040 B
    constexpr int MT = 64;                  // query rows per block
    __shared__ unsigned short hS[MT * LDH]; // ~66.6 KB
    __shared__ float red[8][MT];            // 2 KB

    const int tid  = threadIdx.x;
    const int wave = tid >> 5;
    const int lane = tid & 31;
    const int q0   = blockIdx.x * MT;
    const int w    = blockIdx.y;

    // ---- phase 1: build h tile in LDS (float4 loads, packed bf16 stores) ----
    const float* pw = ppart + (size_t)w * H;
    for (int i = tid; i < MT * H / 4; i += 256) {
        const int idx = i * 4;
        const int row = idx >> 9;      // /512
        const int col = idx & (H - 1); // %512
        const float4 q = *(const float4*)(qpart + (size_t)(q0 + row) * H + col);
        const float4 p = *(const float4*)(pw + col);
        const float v0 = fmaxf(q.x + p.x, 0.f);
        const float v1 = fmaxf(q.y + p.y, 0.f);
        const float v2 = fmaxf(q.z + p.z, 0.f);
        const float v3 = fmaxf(q.w + p.w, 0.f);
        const unsigned int lo = (unsigned)f2bf(v0) | ((unsigned)f2bf(v1) << 16);
        const unsigned int hi = (unsigned)f2bf(v2) | ((unsigned)f2bf(v3) << 16);
        *(uint2*)(hS + row * LDH + col) = make_uint2(lo, hi);
    }
    __syncthreads();

    // ---- phase 2: h2 = relu(h @ R2w + R2b), wave owns N strip [n0w, n0w+64) ----
    const int n0w = wave * 64;
    const int ml = lane & 15, hf = lane >> 4;
    v8f acc[4][4];
#pragma unroll
    for (int mt = 0; mt < 4; ++mt)
#pragma unroll
        for (int t = 0; t < 4; ++t) acc[mt][t] = v8f{};

    for (int k0 = 0; k0 < H; k0 += 32) {
        v16bf a[4];
#pragma unroll
        for (int mt = 0; mt < 4; ++mt) {
            ABFrag fa;
            const unsigned short* pa = hS + (mt * 16 + ml) * LDH + k0 + 8 * hf;
            fa.u[0] = *(const uint4*)(pa);        // ds_read_b128
            fa.u[1] = *(const uint4*)(pa + 16);   // ds_read_b128
            a[mt] = fa.v;
        }
#pragma unroll
        for (int t = 0; t < 4; ++t) {
            v16bf b = load_frag_b(R2wT, H, n0w + 16 * t, k0, lane);
#pragma unroll
            for (int mt = 0; mt < 4; ++mt)
                acc[mt][t] = __builtin_amdgcn_wmma_f32_16x16x32_bf16(
                    false, a[mt], false, b, (short)0, acc[mt][t], false, false);
        }
    }

    // ---- phase 3: fused epilogue + dot with R3w ----
    float partial[4][8];
#pragma unroll
    for (int mt = 0; mt < 4; ++mt)
#pragma unroll
        for (int r = 0; r < 8; ++r) partial[mt][r] = 0.f;
#pragma unroll
    for (int t = 0; t < 4; ++t) {
        const int n = n0w + 16 * t + ml;
        const float bv = R2b[n];
        const float r3 = R3w[n];
#pragma unroll
        for (int mt = 0; mt < 4; ++mt)
#pragma unroll
            for (int r = 0; r < 8; ++r) {
                const float v = fmaxf(acc[mt][t][r] + bv, 0.f);
                partial[mt][r] += v * r3;
            }
    }
    // reduce across the 16 lanes of each half-wave (rows differ per half)
#pragma unroll
    for (int off = 1; off < 16; off <<= 1) {
#pragma unroll
        for (int mt = 0; mt < 4; ++mt)
#pragma unroll
            for (int r = 0; r < 8; ++r)
                partial[mt][r] += __shfl_xor(partial[mt][r], off, 32);
    }
    if ((lane & 15) == 0) {
#pragma unroll
        for (int mt = 0; mt < 4; ++mt)
#pragma unroll
            for (int r = 0; r < 8; ++r)
                red[wave][mt * 16 + r + 8 * hf] = partial[mt][r];
    }
    __syncthreads();

    // ---- phase 4: cross-wave reduce (fixed order => deterministic), sigmoid ----
    if (tid < MT) {
        float s = 0.f;
#pragma unroll
        for (int v = 0; v < 8; ++v) s += red[v][tid];
        s += R3b[0];
        out[(size_t)(q0 + tid) * NUM_WAYS + w] = 1.0f / (1.0f + __expf(-s));
    }
}

// ---------------------------------------------------------------------------
// Host launcher
// ---------------------------------------------------------------------------
extern "C" void kernel_launch(void* const* d_in, const int* in_sizes, int n_in,
                              void* d_out, int out_size, void* d_ws, size_t ws_size,
                              hipStream_t stream) {
    const float* support_x = (const float*)d_in[0];
    const int*   support_y = (const int*)  d_in[1];
    const float* query_x   = (const float*)d_in[2];
    const float* W1  = (const float*)d_in[3];
    const float* b1  = (const float*)d_in[4];
    const float* W2  = (const float*)d_in[5];
    const float* b2  = (const float*)d_in[6];
    const float* R1w = (const float*)d_in[7];
    const float* R1b = (const float*)d_in[8];
    const float* R2w = (const float*)d_in[9];
    const float* R2b = (const float*)d_in[10];
    const float* R3w = (const float*)d_in[11];
    const float* R3b = (const float*)d_in[12];
    float* out = (float*)d_out;

    char* ws = (char*)d_ws;
    size_t off = 0;
    auto carve = [&](size_t bytes) -> char* {
        char* p = ws + off;
        off += (bytes + 255) & ~(size_t)255;
        return p;
    };

    unsigned short* sxb  = (unsigned short*)carve((size_t)NUM_SUP * INPUT_DIM * 2);
    unsigned short* qxb  = (unsigned short*)carve((size_t)NUM_QRY * INPUT_DIM * 2);
    unsigned short* w1t  = (unsigned short*)carve((size_t)HIDDEN * INPUT_DIM * 2);
    unsigned short* w2t  = (unsigned short*)carve((size_t)HIDDEN * HIDDEN * 2);
    unsigned short* r1at = (unsigned short*)carve((size_t)HIDDEN * HIDDEN * 2);
    unsigned short* r1bt = (unsigned short*)carve((size_t)HIDDEN * HIDDEN * 2);
    unsigned short* r2wt = (unsigned short*)carve((size_t)HIDDEN * HIDDEN * 2);
    unsigned short* hsb  = (unsigned short*)carve((size_t)NUM_SUP * HIDDEN * 2);
    unsigned short* sfb  = (unsigned short*)carve((size_t)NUM_SUP * HIDDEN * 2);
    unsigned short* hqb  = (unsigned short*)carve((size_t)NUM_QRY * HIDDEN * 2);
    unsigned short* qfb  = (unsigned short*)carve((size_t)NUM_QRY * HIDDEN * 2);
    float*          qpart   = (float*)carve((size_t)NUM_QRY * HIDDEN * 4);
    float*          protos  = (float*)carve((size_t)NUM_WAYS * HIDDEN * 4);
    unsigned short* protosb = (unsigned short*)carve((size_t)NUM_WAYS * HIDDEN * 2);
    float*          ppart   = (float*)carve((size_t)NUM_WAYS * HIDDEN * 4);
    (void)ws_size; (void)in_sizes; (void)n_in; (void)out_size;

    // 1) activations -> bf16
    {
        int n = NUM_SUP * INPUT_DIM;
        cvt_f32_to_bf16<<<(n + 255) / 256, 256, 0, stream>>>(support_x, sxb, n);
        n = NUM_QRY * INPUT_DIM;
        cvt_f32_to_bf16<<<(n + 255) / 256, 256, 0, stream>>>(query_x, qxb, n);
    }
    // 2) weights -> transposed bf16 [N,K]
    {
        int n = INPUT_DIM * HIDDEN;
        transpose_cvt<<<(n + 255) / 256, 256, 0, stream>>>(W1, w1t, INPUT_DIM, HIDDEN);
        n = HIDDEN * HIDDEN;
        transpose_cvt<<<(n + 255) / 256, 256, 0, stream>>>(W2, w2t, HIDDEN, HIDDEN);
        transpose_cvt<<<(n + 255) / 256, 256, 0, stream>>>(R1w, r1at, HIDDEN, HIDDEN);
        transpose_cvt<<<(n + 255) / 256, 256, 0, stream>>>(R1w + (size_t)HIDDEN * HIDDEN,
                                                           r1bt, HIDDEN, HIDDEN);
        transpose_cvt<<<(n + 255) / 256, 256, 0, stream>>>(R2w, r2wt, HIDDEN, HIDDEN);
    }
    // 3) encoder (support): relu(x@W1+b1) -> relu(.@W2+b2)
    gemm512<true, true><<<NUM_SUP / 32, 256, 0, stream>>>(sxb, w1t, b1, hsb, INPUT_DIM);
    gemm512<true, true><<<NUM_SUP / 32, 256, 0, stream>>>(hsb, w2t, b2, sfb, HIDDEN);
    // 4) encoder (query)
    gemm512<true, true><<<NUM_QRY / 32, 256, 0, stream>>>(qxb, w1t, b1, hqb, INPUT_DIM);
    gemm512<true, true><<<NUM_QRY / 32, 256, 0, stream>>>(hqb, w2t, b2, qfb, HIDDEN);
    // 5) prototypes (segment mean)
    proto_kernel<<<NUM_WAYS, 256, 0, stream>>>(sfb, support_y, protos, protosb);
    // 6) p_part = protos @ R1w[H:] + R1b   (fp32 out, no relu)
    gemm512<false, false><<<NUM_WAYS / 32, 256, 0, stream>>>(protosb, r1bt, R1b, ppart, HIDDEN);
    // 7) q_part = qf @ R1w[:H]             (fp32 out, no bias, no relu)
    gemm512<false, false><<<NUM_QRY / 32, 256, 0, stream>>>(qfb, r1at, nullptr, qpart, HIDDEN);
    // 8) fused relation: relu(q+p) @ R2w -> relu -> @R3w -> sigmoid
    dim3 grid(NUM_QRY / 64, NUM_WAYS);
    relation_fused<<<grid, 256, 0, stream>>>(qpart, ppart, r2wt, R2b, R3w, R3b, out);
}
